// ASLDVS_SNN_73143293051035
// MI455X (gfx1250) — matrix operand order repaired and checked by reading, not compiled, not run on hardware
//
#include <hip/hip_runtime.h>
#include <hip/hip_bf16.h>
#include <stdint.h>

// ---------------- CDNA5 types ----------------
typedef __attribute__((ext_vector_type(16))) _Float16 v16h;
typedef __attribute__((ext_vector_type(8)))  _Float16 v8h;
typedef __attribute__((ext_vector_type(8)))  float    v8f;
typedef __attribute__((ext_vector_type(4)))  unsigned int u32x4;
typedef __attribute__((ext_vector_type(8)))  unsigned int u32x8v;
typedef __attribute__((ext_vector_type(8)))  int      i32x8;
typedef __attribute__((ext_vector_type(4)))  int      i32x4;

#ifdef __gfx1250__
#if __has_builtin(__builtin_amdgcn_tensor_load_to_lds) && __has_builtin(__builtin_amdgcn_s_wait_tensorcnt)
#define USE_TDM 1
#endif
#endif
#ifndef USE_TDM
#define USE_TDM 0
#endif

// ---------------- problem geometry ----------------
#define NB 64
#define NT 64
#define H0 45
#define W0 60
#define H1 23
#define W1 30
#define P1 (H1*W1)            // 690
#define H2 12
#define W2 15
#define P2 (H2*W2)            // 180
#define H3 6
#define W3 8
#define P3 (H3*W3)            // 48
#define C1 16
#define C2 32
#define C3 64
#define FC 128
#define NC 24
#define K1 50
#define K1P 64
#define K2 144
#define K2P 160
#define K3 288
#define FRAME_ELEMS (2*H0*W0) // 5400
#define FRAME_BYTES (FRAME_ELEMS*4)

// halo-padded, channel-PAIR-interleaved activation planes (u32 = 2 f16 chans)
#define FH_W   64              // 60+4
#define FH_U32 (49*FH_W)       // 3136 u32 (both input channels interleaved)
#define S1_W   32              // 30+2
#define S1_CH  (25*S1_W)       // 800 u32 per channel-pair plane (8 pairs)
#define S2_W   18              // 15+2 -> 18
#define S2_CH  (14*S2_W)       // 252 u32 per channel-pair plane (16 pairs)

// ---------------- LDS layout (bytes) ~252 KB of 320 KB/WGP ----------------
#define OFF_STAGE  0                      // 2 x 5400 f32       = 43200
#define OFF_HALOF  43200                  // 3136 u32           = 12544
#define OFF_M1     55744                  // 16x690  f32        = 44160
#define OFF_S1I    99904                  // 8x800   u32        = 25600
#define OFF_M2     125504                 // 32x180  f32        = 23040
#define OFF_S2I    148544                 // 16x252  u32        = 16128
#define OFF_M3     164672                 // 64x48   f32        = 12288
#define OFF_S3I    176960                 // 32x48   u32        =  6144
#define OFF_W1R    183104                 // 16x64   f16        =  2048
#define OFF_W2R    185152                 // 32x160  f16        = 10240
#define OFF_W3R    195392                 // 64x288  f16        = 36864
#define OFF_FC1H   232256                 // 128x64  f16        = 16384
#define OFF_FC2H   248640                 // 24x128  f16        =  6144
#define OFF_SC1    254784
#define OFF_SB1    254848
#define OFF_SC2    254912
#define OFF_SB2    255040
#define OFF_SC3    255168
#define OFF_SB3    255424
#define OFF_FC1B   255680
#define OFF_FC2B   256192
#define OFF_POOL   256320
#define OFF_M4     256576
#define OFF_S4     257088
#define OFF_M5     257600
#define OFF_ACC    257728
#define SMEM_BYTES 257856

// ---------------- WMMA helper ----------------
__device__ __forceinline__ v8f wmma32(v16h a, v16h b, v8f c) {
  return __builtin_amdgcn_wmma_f32_16x16x32_f16(
      false, a, false, b, (short)0, c, false, false);
}

// A fragment from ROW-major weights w[m][kpad]: a lane's 16 halves are two
// consecutive 8-half K runs -> exactly 2x ds_load_b128, zero packing.
__device__ __forceinline__ v16h afrag_row(const _Float16* w, int m, int kpad,
                                          int kbase, int lane) {
  const int grp = (lane >= 16) ? 16 : 0;
  const _Float16* p = w + m * kpad + kbase + grp;   // 16B aligned by construction
  const v8h lo = *(const v8h*)(p);
  const v8h hi = *(const v8h*)(p + 8);
  return __builtin_shufflevector(lo, hi, 0, 1, 2, 3, 4, 5, 6, 7,
                                 8, 9, 10, 11, 12, 13, 14, 15);
}

// ---- B fragments: each VGPR = one u32 (two adjacent channels, same pixel) ----
// conv1: K order k = q*2 + ic; VGPR i -> q = 16ks + 8g + i, halves = ic 0/1.
__device__ __forceinline__ v16h b1frag(const unsigned* fi, int ks, int base, int lane) {
  u32x8v r;
  if (lane < 16) {
#pragma unroll
    for (int i = 0; i < 8; ++i) {
      int q = ks * 16 + i; if (q > 24) q = 24;
      const int kh = q / 5, kw = q - kh * 5;
      r[i] = fi[base + kh * FH_W + kw];
    }
  } else {
#pragma unroll
    for (int i = 0; i < 8; ++i) {
      int q = ks * 16 + 8 + i; if (q > 24) q = 24;
      const int kh = q / 5, kw = q - kh * 5;
      r[i] = fi[base + kh * FH_W + kw];
    }
  }
  return __builtin_bit_cast(v16h, r);
}

// conv2: K order k = q*16 + ic; VGPR i = channel pair i at one (kh,kw).
__device__ __forceinline__ v16h b2frag(const unsigned* s1i, int ks, int base, int lane) {
  int q = ks * 2 + ((lane >= 16) ? 1 : 0); if (q > 8) q = 8;
  const int kh = q / 3, kw = q - kh * 3;
  const int a0 = base + kh * S1_W + kw;
  u32x8v r;
#pragma unroll
  for (int i = 0; i < 8; ++i) r[i] = s1i[a0 + i * S1_CH];
  return __builtin_bit_cast(v16h, r);
}

// conv3: K order k = q*32 + ic; q == ks (const), pair base = 8*lanegrp.
__device__ __forceinline__ v16h b3frag(const unsigned* s2i, int ks, int base, int lane) {
  const int kh = ks / 3, kw = ks - kh * 3;          // constants after unroll
  const int pb = (lane >= 16) ? 8 : 0;
  const int a0 = pb * S2_CH + base + kh * S2_W + kw;
  u32x8v r;
#pragma unroll
  for (int i = 0; i < 8; ++i) r[i] = s2i[a0 + i * S2_CH];
  return __builtin_bit_cast(v16h, r);
}

// BN + LIF epilogue; spikes written as packed channel-pairs (f16 1.0 = 0x3C00).
// D layout: VGPR r of lane L -> (M = r + 8*(L>=16), N = L&15); mbase is even.
__device__ __forceinline__ void lif_epilogue(v8f c, int mt, int nt, int lane,
                                             const float* SC, const float* SB,
                                             float* M, unsigned* SI,
                                             int npix, int wdim, int spw,
                                             int schu, int ho) {
  const int nn = nt * 16 + (lane & 15);
  if (nn >= npix) return;
  const int oh = nn / wdim, ow = nn - oh * wdim;
  const int sidx0 = (oh + ho) * spw + (ow + ho);
  const int mbase = mt * 16 + ((lane >= 16) ? 8 : 0);
#pragma unroll
  for (int pr = 0; pr < 4; ++pr) {
    const int oc0 = mbase + 2 * pr;
    unsigned pack = 0u;
    {
      const float cur = c[2 * pr] * SC[oc0] + SB[oc0];
      const float mem = M[oc0 * npix + nn];
      const float reset = (mem > 1.0f) ? 1.0f : 0.0f;
      const float mn = 0.9f * mem + cur - reset;
      M[oc0 * npix + nn] = mn;
      if (mn > 1.0f) pack |= 0x3C00u;
    }
    {
      const int oc1 = oc0 + 1;
      const float cur = c[2 * pr + 1] * SC[oc1] + SB[oc1];
      const float mem = M[oc1 * npix + nn];
      const float reset = (mem > 1.0f) ? 1.0f : 0.0f;
      const float mn = 0.9f * mem + cur - reset;
      M[oc1 * npix + nn] = mn;
      if (mn > 1.0f) pack |= 0x3C000000u;
    }
    SI[(oc0 >> 1) * schu + sidx0] = pack;
  }
}

// ---------------- TDM input staging ----------------
#if USE_TDM
__device__ __forceinline__ void tdm_load_frame(const float* gsrc, unsigned lds_off) {
  const unsigned long long ga = (unsigned long long)(uintptr_t)gsrc;
  u32x4 g0;
  g0[0] = 1u;                                        // count=1
  g0[1] = lds_off;                                   // lds_addr
  g0[2] = (unsigned)(ga & 0xFFFFFFFFu);
  g0[3] = (unsigned)((ga >> 32) & 0x1FFFFFFu) | (2u << 30);  // type=2
  i32x8 g1 = {0, 0, 0, 0, 0, 0, 0, 0};
  g1[0] = (int)(2u << 16);                           // data_size=4B
  g1[1] = (int)((unsigned)FRAME_ELEMS << 16);        // tensor_dim0 lo16
  g1[2] = (int)(1u << 16);                           // dim0 hi=0, tensor_dim1=1
  g1[3] = (int)((unsigned)FRAME_ELEMS << 16);        // tile_dim0
  g1[4] = 1;                                         // tile_dim1=1
  g1[5] = FRAME_ELEMS;                               // dim0 stride
  const i32x4 gz = {0, 0, 0, 0};
#if __clang_major__ >= 23
  const i32x8 gz8 = {0, 0, 0, 0, 0, 0, 0, 0};
  __builtin_amdgcn_tensor_load_to_lds(g0, g1, gz, gz, gz8, 0);
#else
  __builtin_amdgcn_tensor_load_to_lds(g0, g1, gz, gz, 0);
#endif
}
#endif

// raw f32 frame (staging or global) -> channel-pair-interleaved f16 halo frame
__device__ __forceinline__ void frame_convert(const float* src, unsigned* fi, int tid) {
  for (int i = tid; i < H0 * W0; i += 256) {
    const int r = i / W0, cc = i - r * W0;
    const _Float16 h0 = (_Float16)src[i];
    const _Float16 h1 = (_Float16)src[H0 * W0 + i];
    const unsigned p = (unsigned)__builtin_bit_cast(unsigned short, h0)
                     | ((unsigned)__builtin_bit_cast(unsigned short, h1) << 16);
    fi[(r + 2) * FH_W + (cc + 2)] = p;
  }
}

// ---------------- fused SNN kernel: one block == one batch image ----------------
__global__ __launch_bounds__(256, 1)
void asldvs_snn_fused(const float* __restrict__ xg,
                      const float* __restrict__ c1w, const float* __restrict__ c1b,
                      const float* __restrict__ c2w, const float* __restrict__ c2b,
                      const float* __restrict__ c3w, const float* __restrict__ c3b,
                      const float* __restrict__ bn1g, const float* __restrict__ bn1b,
                      const float* __restrict__ bn1m, const float* __restrict__ bn1v,
                      const float* __restrict__ bn2g, const float* __restrict__ bn2b,
                      const float* __restrict__ bn2m, const float* __restrict__ bn2v,
                      const float* __restrict__ bn3g, const float* __restrict__ bn3b,
                      const float* __restrict__ bn3m, const float* __restrict__ bn3v,
                      const float* __restrict__ fc1w, const float* __restrict__ fc1b,
                      const float* __restrict__ fc2w, const float* __restrict__ fc2b,
                      float* __restrict__ out) {
  extern __shared__ char smem[];
  float*    STAGE = (float*)(smem + OFF_STAGE);
  unsigned* FI    = (unsigned*)(smem + OFF_HALOF);
  float*    M1    = (float*)(smem + OFF_M1);
  unsigned* S1I   = (unsigned*)(smem + OFF_S1I);
  float*    M2    = (float*)(smem + OFF_M2);
  unsigned* S2I   = (unsigned*)(smem + OFF_S2I);
  float*    M3    = (float*)(smem + OFF_M3);
  unsigned* S3I   = (unsigned*)(smem + OFF_S3I);
  _Float16* W1R   = (_Float16*)(smem + OFF_W1R);
  _Float16* W2R   = (_Float16*)(smem + OFF_W2R);
  _Float16* W3R   = (_Float16*)(smem + OFF_W3R);
  _Float16* FC1H  = (_Float16*)(smem + OFF_FC1H);
  _Float16* FC2H  = (_Float16*)(smem + OFF_FC2H);
  float* SC1 = (float*)(smem + OFF_SC1); float* SB1 = (float*)(smem + OFF_SB1);
  float* SC2 = (float*)(smem + OFF_SC2); float* SB2 = (float*)(smem + OFF_SB2);
  float* SC3 = (float*)(smem + OFF_SC3); float* SB3 = (float*)(smem + OFF_SB3);
  float* FC1B = (float*)(smem + OFF_FC1B);
  float* FC2B = (float*)(smem + OFF_FC2B);
  float* POOL = (float*)(smem + OFF_POOL);
  float* M4 = (float*)(smem + OFF_M4);
  float* S4 = (float*)(smem + OFF_S4);
  float* M5 = (float*)(smem + OFF_M5);
  float* ACC = (float*)(smem + OFF_ACC);

  const int tid = threadIdx.x;
  const int wid = tid >> 5;
  const int lane = tid & 31;
  const int b = blockIdx.x;

  // --- one-time staging: weights -> row-major [oc][Kpad] in fragment K-order ---
  for (int i = tid; i < C1 * K1P; i += 256) {     // k' = q*2 + ic
    const int oc = i >> 6, kp = i & 63;
    const int ic = kp & 1, q = kp >> 1;
    W1R[i] = (_Float16)((q < 25) ? c1w[oc * K1 + ic * 25 + q] : 0.0f);
  }
  for (int i = tid; i < C2 * K2P; i += 256) {     // k' = q*16 + ic
    const int oc = i / K2P, kp = i - oc * K2P;
    const int q = kp >> 4, ic = kp & 15;
    W2R[i] = (_Float16)((q < 9) ? c2w[oc * K2 + ic * 9 + q] : 0.0f);
  }
  for (int i = tid; i < C3 * K3; i += 256) {      // k' = q*32 + ic
    const int oc = i / K3, kp = i - oc * K3;
    const int q = kp >> 5, ic = kp & 31;
    W3R[i] = (_Float16)c3w[oc * K3 + ic * 9 + q];
  }
  for (int i = tid; i < FC * C3; i += 256) FC1H[i] = (_Float16)fc1w[i];
  for (int i = tid; i < NC * FC; i += 256) FC2H[i] = (_Float16)fc2w[i];
  if (tid < C1) { const float s = bn1g[tid] * rsqrtf(bn1v[tid] + 1e-5f);
                  SC1[tid] = s; SB1[tid] = bn1b[tid] - bn1m[tid] * s + c1b[tid] * s; }
  if (tid < C2) { const float s = bn2g[tid] * rsqrtf(bn2v[tid] + 1e-5f);
                  SC2[tid] = s; SB2[tid] = bn2b[tid] - bn2m[tid] * s + c2b[tid] * s; }
  if (tid < C3) { const float s = bn3g[tid] * rsqrtf(bn3v[tid] + 1e-5f);
                  SC3[tid] = s; SB3[tid] = bn3b[tid] - bn3m[tid] * s + c3b[tid] * s; }
  if (tid < FC) { FC1B[tid] = fc1b[tid]; M4[tid] = 0.0f; S4[tid] = 0.0f; }
  if (tid < NC) { FC2B[tid] = fc2b[tid]; M5[tid] = 0.0f; ACC[tid] = 0.0f; }
  for (int i = tid; i < C1 * P1; i += 256) M1[i] = 0.0f;
  for (int i = tid; i < C2 * P2; i += 256) M2[i] = 0.0f;
  for (int i = tid; i < C3 * P3; i += 256) M3[i] = 0.0f;
  for (int i = tid; i < FH_U32; i += 256) FI[i] = 0u;
  for (int i = tid; i < (C1 / 2) * S1_CH; i += 256) S1I[i] = 0u;
  for (int i = tid; i < (C2 / 2) * S2_CH; i += 256) S2I[i] = 0u;

  const float* xbase = xg + (size_t)b * NT * FRAME_ELEMS;
#if USE_TDM
  if (tid < 32) { tdm_load_frame(xbase, OFF_STAGE); __builtin_amdgcn_s_wait_tensorcnt(0); }
#endif
  __syncthreads();

  // ---------------- recurrent scan over T ----------------
  for (int t = 0; t < NT; ++t) {
#if USE_TDM
    if (t + 1 < NT && tid < 32)
      tdm_load_frame(xbase + (size_t)(t + 1) * FRAME_ELEMS,
                     OFF_STAGE + (unsigned)(((t + 1) & 1) * FRAME_BYTES));
    frame_convert(STAGE + (t & 1) * FRAME_ELEMS, FI, tid);
#else
    frame_convert(xbase + (size_t)t * FRAME_ELEMS, FI, tid);
    if (tid == 0 && t + 1 < NT)
      __builtin_prefetch(xbase + (size_t)(t + 1) * FRAME_ELEMS, 0, 0);
#endif
    __syncthreads();

    // ---- conv1 + bn1 + lif1 : GEMM 16 x 690 x 64, 44 N-tiles / 8 waves ----
    for (int nt = wid; nt < 44; nt += 8) {
      const int n = nt * 16 + (lane & 15);
      const int nn = (n < P1) ? n : (P1 - 1);
      const int oh = nn / W1, ow = nn - oh * W1;
      const int base = oh * (2 * FH_W) + 2 * ow;     // u32 units, halo coords
      v8f c = {};
#pragma unroll
      for (int ks = 0; ks < 2; ++ks) {
        const v16h a = afrag_row(W1R, lane & 15, K1P, ks * 32, lane);
        const v16h bb = b1frag(FI, ks, base, lane);
        c = wmma32(a, bb, c);
      }
      lif_epilogue(c, 0, nt, lane, SC1, SB1, M1, S1I, P1, W1, S1_W, S1_CH, 1);
    }
    __syncthreads();

    // ---- conv2 + bn2 + lif2 : GEMM 32 x 180 x 160, 2x12 tile pairs ----
    for (int p = wid; p < 24; p += 8) {
      const int mt = p / 12, ntile = p - mt * 12;
      const int n = ntile * 16 + (lane & 15);
      const int nn = (n < P2) ? n : (P2 - 1);
      const int oh = nn / W2, ow = nn - oh * W2;
      const int base = oh * (2 * S1_W) + 2 * ow;
      v8f c = {};
#pragma unroll
      for (int ks = 0; ks < 5; ++ks) {
        const v16h a = afrag_row(W2R, mt * 16 + (lane & 15), K2P, ks * 32, lane);
        const v16h bb = b2frag(S1I, ks, base, lane);
        c = wmma32(a, bb, c);
      }
      lif_epilogue(c, mt, ntile, lane, SC2, SB2, M2, S2I, P2, W2, S2_W, S2_CH, 1);
    }
    __syncthreads();

    // ---- conv3 + bn3 + lif3 : GEMM 64 x 48 x 288, 4x3 tile pairs ----
    for (int p = wid; p < 12; p += 8) {
      const int mt = p / 3, ntile = p - mt * 3;
      const int n = ntile * 16 + (lane & 15);       // n < 48 always valid
      const int oh = n / W3, ow = n - oh * W3;
      const int base = oh * (2 * S2_W) + 2 * ow;
      v8f c = {};
#pragma unroll
      for (int ks = 0; ks < 9; ++ks) {
        const v16h a = afrag_row(W3R, mt * 16 + (lane & 15), K3, ks * 32, lane);
        const v16h bb = b3frag(S2I, ks, base, lane);
        c = wmma32(a, bb, c);
      }
      lif_epilogue(c, mt, ntile, lane, SC3, SB3, M3, S3I, P3, W3, W3, P3, 0);
    }
    __syncthreads();

    // ---- adaptive avg pool: one thread per channel pair, 32B vector reads ----
    if (tid < C3 / 2) {
      const u32x8v* pp = (const u32x8v*)(S3I + tid * P3);
      float s0 = 0.0f, s1 = 0.0f;
#pragma unroll
      for (int ck = 0; ck < 6; ++ck) {
        const v16h v = __builtin_bit_cast(v16h, pp[ck]);
#pragma unroll
        for (int e = 0; e < 16; e += 2) { s0 += (float)v[e]; s1 += (float)v[e + 1]; }
      }
      POOL[2 * tid]     = s0 * (1.0f / (float)P3);
      POOL[2 * tid + 1] = s1 * (1.0f / (float)P3);
    }
    __syncthreads();

    // ---- fc1 + lif4 ----
    if (tid < FC) {
      const v16h* row = (const v16h*)(FC1H + tid * C3);
      float cur = FC1B[tid];
#pragma unroll
      for (int c16 = 0; c16 < 4; ++c16) {
        const v16h w = row[c16];
#pragma unroll
        for (int e = 0; e < 16; ++e) cur += (float)w[e] * POOL[c16 * 16 + e];
      }
      const float mem = M4[tid];
      const float reset = (mem > 1.0f) ? 1.0f : 0.0f;
      const float mn = 0.9f * mem + cur - reset;
      M4[tid] = mn;
      S4[tid] = (mn > 1.0f) ? 1.0f : 0.0f;
    }
    __syncthreads();

    // ---- fc2 + lif5 + membrane accumulation ----
    if (tid < NC) {
      const v16h* row = (const v16h*)(FC2H + tid * FC);
      float cur = FC2B[tid];
#pragma unroll
      for (int c16 = 0; c16 < 8; ++c16) {
        const v16h w = row[c16];
#pragma unroll
        for (int e = 0; e < 16; ++e) cur += (float)w[e] * S4[c16 * 16 + e];
      }
      const float mem = M5[tid];
      const float reset = (mem > 1.0f) ? 1.0f : 0.0f;
      const float mn = 0.9f * mem + cur - reset;
      M5[tid] = mn;
      ACC[tid] += mn;
    }

#if USE_TDM
    if (tid < 32) __builtin_amdgcn_s_wait_tensorcnt(0);
#endif
    __syncthreads();
  }

  if (tid < NC) out[(size_t)b * NC + tid] = ACC[tid];
}

// ---------------- launcher ----------------
extern "C" void kernel_launch(void* const* d_in, const int* in_sizes, int n_in,
                              void* d_out, int out_size, void* d_ws, size_t ws_size,
                              hipStream_t stream) {
  (void)in_sizes; (void)n_in; (void)out_size; (void)d_ws; (void)ws_size;
  const float* x    = (const float*)d_in[0];
  const float* c1w  = (const float*)d_in[1];
  const float* c1b  = (const float*)d_in[2];
  const float* c2w  = (const float*)d_in[3];
  const float* c2b  = (const float*)d_in[4];
  const float* c3w  = (const float*)d_in[5];
  const float* c3b  = (const float*)d_in[6];
  const float* bn1g = (const float*)d_in[7];
  const float* bn1b = (const float*)d_in[8];
  const float* bn1m = (const float*)d_in[9];
  const float* bn1v = (const float*)d_in[10];
  const float* bn2g = (const float*)d_in[11];
  const float* bn2b = (const float*)d_in[12];
  const float* bn2m = (const float*)d_in[13];
  const float* bn2v = (const float*)d_in[14];
  const float* bn3g = (const float*)d_in[15];
  const float* bn3b = (const float*)d_in[16];
  const float* bn3m = (const float*)d_in[17];
  const float* bn3v = (const float*)d_in[18];
  const float* fc1w = (const float*)d_in[19];
  const float* fc1b = (const float*)d_in[20];
  const float* fc2w = (const float*)d_in[21];
  const float* fc2b = (const float*)d_in[22];

  asldvs_snn_fused<<<dim3(NB), dim3(256), SMEM_BYTES, stream>>>(
      x, c1w, c1b, c2w, c2b, c3w, c3b,
      bn1g, bn1b, bn1m, bn1v, bn2g, bn2b, bn2m, bn2v, bn3g, bn3b, bn3m, bn3v,
      fc1w, fc1b, fc2w, fc2b, (float*)d_out);
}
